// Generator_35880156790836
// MI455X (gfx1250) — matrix operand order repaired and checked by reading, not compiled, not run on hardware
//
#include <hip/hip_runtime.h>
#include <hip/hip_bf16.h>

// ---------------------------------------------------------------------------
// Types for CDNA5 WMMA (wave32): v_wmma_f32_16x16x32_bf16
// ---------------------------------------------------------------------------
typedef __attribute__((ext_vector_type(8)))  float  v8f;
typedef __attribute__((ext_vector_type(16))) __bf16 v16bf;
typedef __attribute__((ext_vector_type(8)))  __bf16 v8bf;

union Frag16 { v16bf v; v8bf h[2]; };

#define BATCH    64
#define LATENT   128
#define WEMBED   256
#define HID      1024
#define VOCAB    32000
#define SEQ      64

__device__ __forceinline__ unsigned short f2bf(float f) {
    union { float f; unsigned int u; } x; x.f = f;
    unsigned int u = x.u;
    unsigned int r = u + 0x7FFFu + ((u >> 16) & 1u);   // round-to-nearest-even
    return (unsigned short)(r >> 16);
}

// ---------------------------------------------------------------------------
// Generic bf16 WMMA GEMM:  C[64,N] = X[64,K] * W[N,K]^T (+ bias)
// One wave computes a 64(M) x 16(N) strip: the B (weight) fragment is loaded
// once per K-chunk and reused for 4 A fragments -> each weight byte is read
// exactly once per GEMM (weights live in the 192MB L2 as bf16).
// Software-pipelined with two fragment buffers: loads for chunk i+1 are
// issued before the WMMAs of chunk i, so the per-wave loadcnt waits overlap
// with matrix ops instead of serializing on loadcnt==0.
// Requires K to be a multiple of 64 (384 / 1024 / 32000 all qualify).
// ---------------------------------------------------------------------------
struct ChunkFrags { Frag16 b; Frag16 a[4]; };

__device__ __forceinline__ void load_chunk(ChunkFrags& f,
                                           const unsigned short* __restrict__ wrow,
                                           const unsigned short* __restrict__ arow0,
                                           const unsigned short* __restrict__ arow1,
                                           const unsigned short* __restrict__ arow2,
                                           const unsigned short* __restrict__ arow3,
                                           int kb) {
    f.b.h[0] = *(const v8bf*)(wrow + kb);
    f.b.h[1] = *(const v8bf*)(wrow + kb + 16);
    f.a[0].h[0] = *(const v8bf*)(arow0 + kb);
    f.a[0].h[1] = *(const v8bf*)(arow0 + kb + 16);
    f.a[1].h[0] = *(const v8bf*)(arow1 + kb);
    f.a[1].h[1] = *(const v8bf*)(arow1 + kb + 16);
    f.a[2].h[0] = *(const v8bf*)(arow2 + kb);
    f.a[2].h[1] = *(const v8bf*)(arow2 + kb + 16);
    f.a[3].h[0] = *(const v8bf*)(arow3 + kb);
    f.a[3].h[1] = *(const v8bf*)(arow3 + kb + 16);
}

__device__ __forceinline__ void wmma4(v8f acc[4], const ChunkFrags& f) {
    #pragma unroll
    for (int mt = 0; mt < 4; ++mt) {
        acc[mt] = __builtin_amdgcn_wmma_f32_16x16x32_bf16(
                      false, f.a[mt].v, false, f.b.v,
                      (short)0, acc[mt], false, false);
    }
}

__global__ __launch_bounds__(256)
void wmma_gemm_bf16(const unsigned short* __restrict__ X,
                    const unsigned short* __restrict__ W,
                    const float* __restrict__ bias,
                    float* __restrict__ C,
                    int K, int N) {
    int wave = (int)((blockIdx.x * blockDim.x + threadIdx.x) >> 5);
    int n0 = wave * 16;
    if (n0 >= N) return;

    int lane = (int)(threadIdx.x & 31);
    int half = lane >> 4;      // 0: lanes 0-15, 1: lanes 16-31
    int l15  = lane & 15;

    v8f acc[4] = {};           // 4 M-tiles of 16x16 f32 accumulators

    // B fragment source: lane holds N = n0+l15,
    //   K = kb + 8*half + e      (e in [0,8))
    //   K = kb + 8*half + 8 + e  (e in [8,16))
    const unsigned short* wrow = W + (size_t)(n0 + l15) * (size_t)K + half * 8;
    // A fragment sources: lane holds M = mt*16+l15,
    //   K = kb + 8*half + e      (e in [0,8))
    //   K = kb + 8*half + 8 + e  (e in [8,16))
    const unsigned short* arow0 = X + (size_t)( 0 + l15) * (size_t)K + half * 8;
    const unsigned short* arow1 = X + (size_t)(16 + l15) * (size_t)K + half * 8;
    const unsigned short* arow2 = X + (size_t)(32 + l15) * (size_t)K + half * 8;
    const unsigned short* arow3 = X + (size_t)(48 + l15) * (size_t)K + half * 8;

    // Software pipeline: two chunk buffers, K is a multiple of 64.
    ChunkFrags f0, f1;
    load_chunk(f0, wrow, arow0, arow1, arow2, arow3, 0);
    for (int kb = 0; kb < K - 64; kb += 64) {
        load_chunk(f1, wrow, arow0, arow1, arow2, arow3, kb + 32);
        wmma4(acc, f0);
        load_chunk(f0, wrow, arow0, arow1, arow2, arow3, kb + 64);
        wmma4(acc, f1);
    }
    load_chunk(f1, wrow, arow0, arow1, arow2, arow3, K - 32);
    wmma4(acc, f0);
    wmma4(acc, f1);

    float bvv = bias ? bias[n0 + l15] : 0.0f;
    #pragma unroll
    for (int mt = 0; mt < 4; ++mt) {
        #pragma unroll
        for (int r = 0; r < 8; ++r) {
            int m = mt * 16 + r + half * 8;   // C/D layout: reg r, half-group
            C[(size_t)m * (size_t)N + (size_t)(n0 + l15)] = acc[mt][r] + bvv;
        }
    }
}

// ---------------------------------------------------------------------------
// Prep kernels
// ---------------------------------------------------------------------------
__global__ void convert_bf16(const float* __restrict__ src,
                             unsigned short* __restrict__ dst, size_t n) {
    size_t i = (size_t)blockIdx.x * blockDim.x + threadIdx.x;
    if (i < n) dst[i] = f2bf(src[i]);
}

// embT[c][v] = bf16(emb[v][c])  -> weight layout [N=256, K=32000]
__global__ void transpose_emb(const float* __restrict__ emb,
                              unsigned short* __restrict__ embT) {
    size_t i = (size_t)blockIdx.x * blockDim.x + threadIdx.x;
    if (i >= (size_t)VOCAB * WEMBED) return;
    size_t v = i / WEMBED, c = i % WEMBED;
    embT[c * (size_t)VOCAB + v] = f2bf(emb[i]);
}

// x0 = [zeros(64,256), z] as bf16
__global__ void init_x(const float* __restrict__ z,
                       unsigned short* __restrict__ xbuf) {
    int i = blockIdx.x * blockDim.x + threadIdx.x;
    if (i >= BATCH * (WEMBED + LATENT)) return;
    int b = i / (WEMBED + LATENT), c = i % (WEMBED + LATENT);
    float v = (c < WEMBED) ? 0.0f : z[b * LATENT + (c - WEMBED)];
    xbuf[i] = f2bf(v);
}

// h0 = elu(z @ Wl^T + bl), tiny GEMM (K=128), f32 + bf16 copies
__global__ void init_h(const float* __restrict__ z,
                       const float* __restrict__ Wl,
                       const float* __restrict__ bl,
                       float* __restrict__ hf,
                       unsigned short* __restrict__ hbf) {
    int b = blockIdx.x;
    for (int n = threadIdx.x; n < HID; n += (int)blockDim.x) {
        float s = bl[n];
        const float* zr = z + b * LATENT;
        const float* wr = Wl + (size_t)n * LATENT;
        for (int k = 0; k < LATENT; ++k) s += zr[k] * wr[k];
        float e = (s > 0.0f) ? s : (__expf(s) - 1.0f);
        hf[b * HID + n] = e;
        hbf[b * HID + n] = f2bf(e);
    }
}

// ---------------------------------------------------------------------------
// Fused LayerNorm + GRU cell update (block per batch row)
//   gi = LN(x@Wx^T + bx), gh = LN(h@Wh^T + bh)  (pre-activations in gi/gh)
// ---------------------------------------------------------------------------
__global__ __launch_bounds__(256)
void ln_gru(const float* __restrict__ gi, const float* __restrict__ gh,
            const float* __restrict__ g_i, const float* __restrict__ b_i,
            const float* __restrict__ g_h, const float* __restrict__ b_h,
            float* __restrict__ h, unsigned short* __restrict__ hbf) {
    const int H3 = 3 * HID;
    int b = blockIdx.x;
    const float* gir = gi + (size_t)b * H3;
    const float* ghr = gh + (size_t)b * H3;
    __shared__ float s0[256], s1[256];
    int tid = threadIdx.x;

    // stats of gi row
    float sum = 0.0f, sq = 0.0f;
    for (int j = tid; j < H3; j += 256) { float v = gir[j]; sum += v; sq += v * v; }
    s0[tid] = sum; s1[tid] = sq; __syncthreads();
    for (int off = 128; off > 0; off >>= 1) {
        if (tid < off) { s0[tid] += s0[tid + off]; s1[tid] += s1[tid + off]; }
        __syncthreads();
    }
    float mu_i = s0[0] / (float)H3;
    float ri   = rsqrtf(s1[0] / (float)H3 - mu_i * mu_i + 1e-5f);
    __syncthreads();

    // stats of gh row
    sum = 0.0f; sq = 0.0f;
    for (int j = tid; j < H3; j += 256) { float v = ghr[j]; sum += v; sq += v * v; }
    s0[tid] = sum; s1[tid] = sq; __syncthreads();
    for (int off = 128; off > 0; off >>= 1) {
        if (tid < off) { s0[tid] += s0[tid + off]; s1[tid] += s1[tid + off]; }
        __syncthreads();
    }
    float mu_h = s0[0] / (float)H3;
    float rh   = rsqrtf(s1[0] / (float)H3 - mu_h * mu_h + 1e-5f);

    for (int j = tid; j < HID; j += 256) {
        float irv = (gir[j]          - mu_i) * ri * g_i[j]          + b_i[j];
        float izv = (gir[j + HID]    - mu_i) * ri * g_i[j + HID]    + b_i[j + HID];
        float inv = (gir[j + 2*HID]  - mu_i) * ri * g_i[j + 2*HID]  + b_i[j + 2*HID];
        float hrv = (ghr[j]          - mu_h) * rh * g_h[j]          + b_h[j];
        float hzv = (ghr[j + HID]    - mu_h) * rh * g_h[j + HID]    + b_h[j + HID];
        float hnv = (ghr[j + 2*HID]  - mu_h) * rh * g_h[j + 2*HID]  + b_h[j + 2*HID];

        float r  = 1.0f / (1.0f + __expf(-(irv + hrv)));
        float zg = 1.0f / (1.0f + __expf(-(izv + hzv)));
        float n  = tanhf(inv + r * hnv);
        float hv = h[(size_t)b * HID + j];
        float hn = (1.0f - zg) * n + zg * hv;
        h[(size_t)b * HID + j] = hn;
        hbf[(size_t)b * HID + j] = f2bf(hn);
    }
}

// ---------------------------------------------------------------------------
// softmax(logits / 0.001) -> probs (bf16), block per batch row
// ---------------------------------------------------------------------------
__global__ __launch_bounds__(256)
void softmax_probs(const float* __restrict__ logits,
                   unsigned short* __restrict__ probs) {
    int b = blockIdx.x;
    const float* L = logits + (size_t)b * VOCAB;
    unsigned short* P = probs + (size_t)b * VOCAB;
    __shared__ float sred[256];
    int tid = threadIdx.x;

    float mx = -3.4e38f;
    for (int j = tid; j < VOCAB; j += 256) mx = fmaxf(mx, L[j]);
    sred[tid] = mx; __syncthreads();
    for (int off = 128; off > 0; off >>= 1) {
        if (tid < off) sred[tid] = fmaxf(sred[tid], sred[tid + off]);
        __syncthreads();
    }
    mx = sred[0]; __syncthreads();

    float sum = 0.0f;
    for (int j = tid; j < VOCAB; j += 256) sum += __expf((L[j] - mx) * 1000.0f);
    sred[tid] = sum; __syncthreads();
    for (int off = 128; off > 0; off >>= 1) {
        if (tid < off) sred[tid] += sred[tid + off];
        __syncthreads();
    }
    float inv = 1.0f / sred[0];

    for (int j = tid; j < VOCAB; j += 256)
        P[j] = f2bf(__expf((L[j] - mx) * 1000.0f) * inv);
}

// ---------------------------------------------------------------------------
// write w_emb into d_out[b, t, :] (f32) and refresh x buffer (bf16)
// ---------------------------------------------------------------------------
__global__ void finish_step(const float* __restrict__ wemb,
                            float* __restrict__ out,
                            unsigned short* __restrict__ xbuf, int t) {
    int i = blockIdx.x * blockDim.x + threadIdx.x;
    if (i >= BATCH * WEMBED) return;
    int b = i >> 8, c = i & 255;
    float v = wemb[i];
    out[((size_t)b * SEQ + t) * WEMBED + c] = v;
    xbuf[b * (WEMBED + LATENT) + c] = f2bf(v);
}

// ---------------------------------------------------------------------------
// Host orchestration
// ---------------------------------------------------------------------------
static inline void launch_gemm(const unsigned short* X, const unsigned short* W,
                               const float* bias, float* C, int K, int N,
                               hipStream_t s) {
    int waves = N / 16;                 // N is always a multiple of 16 here
    int blocks = (waves + 7) / 8;       // 8 waves per 256-thread block
    wmma_gemm_bf16<<<blocks, 256, 0, s>>>(X, W, bias, C, K, N);
}

extern "C" void kernel_launch(void* const* d_in, const int* in_sizes, int n_in,
                              void* d_out, int out_size, void* d_ws, size_t ws_size,
                              hipStream_t stream) {
    (void)in_sizes; (void)n_in; (void)out_size; (void)ws_size;

    const float* z    = (const float*)d_in[0];
    const float* emb  = (const float*)d_in[1];
    const float* Wl0  = (const float*)d_in[2];
    const float* bl0  = (const float*)d_in[3];
    const float* Wx0  = (const float*)d_in[4];
    const float* bxc0 = (const float*)d_in[5];
    const float* Wh0  = (const float*)d_in[6];
    const float* bhc0 = (const float*)d_in[7];
    const float* g_i0 = (const float*)d_in[8];
    const float* b_i0 = (const float*)d_in[9];
    const float* g_h0 = (const float*)d_in[10];
    const float* b_h0 = (const float*)d_in[11];
    const float* Wl1  = (const float*)d_in[12];
    const float* bl1  = (const float*)d_in[13];
    const float* Wx1  = (const float*)d_in[14];
    const float* bxc1 = (const float*)d_in[15];
    const float* Wh1  = (const float*)d_in[16];
    const float* bhc1 = (const float*)d_in[17];
    const float* g_i1 = (const float*)d_in[18];
    const float* b_i1 = (const float*)d_in[19];
    const float* g_h1 = (const float*)d_in[20];
    const float* b_h1 = (const float*)d_in[21];
    const float* Wv   = (const float*)d_in[22];
    const float* bv   = (const float*)d_in[23];
    float* out = (float*)d_out;

    // ---- carve scratch (256B-aligned sub-allocations) ----
    char* p = (char*)d_ws;
    auto carve = [&](size_t bytes) -> char* {
        char* r = p; p += (bytes + 255) & ~(size_t)255; return r;
    };
    const int IN0 = WEMBED + LATENT;             // 384
    unsigned short* Wx0b  = (unsigned short*)carve((size_t)3*HID*IN0   * 2);
    unsigned short* Wh0b  = (unsigned short*)carve((size_t)3*HID*HID   * 2);
    unsigned short* Wx1b  = (unsigned short*)carve((size_t)3*HID*HID   * 2);
    unsigned short* Wh1b  = (unsigned short*)carve((size_t)3*HID*HID   * 2);
    unsigned short* Wvb   = (unsigned short*)carve((size_t)VOCAB*HID   * 2);
    unsigned short* embT  = (unsigned short*)carve((size_t)WEMBED*VOCAB* 2);
    unsigned short* probs = (unsigned short*)carve((size_t)BATCH*VOCAB * 2);
    unsigned short* xbuf  = (unsigned short*)carve((size_t)BATCH*IN0   * 2);
    unsigned short* h0b   = (unsigned short*)carve((size_t)BATCH*HID   * 2);
    unsigned short* h1b   = (unsigned short*)carve((size_t)BATCH*HID   * 2);
    float* h0f    = (float*)carve((size_t)BATCH*HID   * 4);
    float* h1f    = (float*)carve((size_t)BATCH*HID   * 4);
    float* gi     = (float*)carve((size_t)BATCH*3*HID * 4);
    float* gh     = (float*)carve((size_t)BATCH*3*HID * 4);
    float* logits = (float*)carve((size_t)BATCH*VOCAB * 4);
    float* wemb   = (float*)carve((size_t)BATCH*WEMBED* 4);

    // ---- one-time prep: bf16 weights (fits in 192MB L2), emb^T, x0, h0 ----
    auto conv = [&](const float* src, unsigned short* dst, size_t n) {
        convert_bf16<<<(unsigned)((n + 255) / 256), 256, 0, stream>>>(src, dst, n);
    };
    conv(Wx0, Wx0b, (size_t)3*HID*IN0);
    conv(Wh0, Wh0b, (size_t)3*HID*HID);
    conv(Wx1, Wx1b, (size_t)3*HID*HID);
    conv(Wh1, Wh1b, (size_t)3*HID*HID);
    conv(Wv,  Wvb,  (size_t)VOCAB*HID);
    transpose_emb<<<(unsigned)(((size_t)VOCAB*WEMBED + 255) / 256), 256, 0, stream>>>(emb, embT);
    init_x<<<(BATCH*IN0 + 255) / 256, 256, 0, stream>>>(z, xbuf);
    init_h<<<BATCH, 256, 0, stream>>>(z, Wl0, bl0, h0f, h0b);
    init_h<<<BATCH, 256, 0, stream>>>(z, Wl1, bl1, h1f, h1b);

    // ---- recurrence ----
    for (int t = 0; t < SEQ; ++t) {
        // layer 0
        launch_gemm(xbuf, Wx0b, bxc0, gi, IN0, 3*HID, stream);
        launch_gemm(h0b,  Wh0b, bhc0, gh, HID, 3*HID, stream);
        ln_gru<<<BATCH, 256, 0, stream>>>(gi, gh, g_i0, b_i0, g_h0, b_h0, h0f, h0b);
        // layer 1 (input = h0)
        launch_gemm(h0b, Wx1b, bxc1, gi, HID, 3*HID, stream);
        launch_gemm(h1b, Wh1b, bhc1, gh, HID, 3*HID, stream);
        ln_gru<<<BATCH, 256, 0, stream>>>(gi, gh, g_i1, b_i1, g_h1, b_h1, h1f, h1b);
        // vocab projection + soft-argmax + weighted embedding
        launch_gemm(h1b, Wvb, bv, logits, HID, VOCAB, stream);
        softmax_probs<<<BATCH, 256, 0, stream>>>(logits, probs);
        launch_gemm(probs, embT, nullptr, wemb, VOCAB, WEMBED, stream);
        finish_step<<<(BATCH*WEMBED + 255) / 256, 256, 0, stream>>>(wemb, out, xbuf, t);
    }
}
